// NestedMIL_57664230916715
// MI455X (gfx1250) — matrix operand order repaired and checked by reading, not compiled, not run on hardware
//
#include <hip/hip_runtime.h>
#include <hip/hip_bf16.h>
#include <math.h>

// ---------------- CDNA5 WMMA types ----------------
typedef __attribute__((ext_vector_type(16))) _Float16 v16h;
typedef __attribute__((ext_vector_type(8)))  float    v8f;

union V16H { v16h v; uint4 q[2]; };

#define NB  32      // batch
#define ND  32      // dims
#define NS  512     // sub-windows (== WIN, SUB=1)
#define NR  (NB*NS)         // 16384 rows (b,s)
#define NM3 (NR*ND)         // 524288 rows (b,s,d)
#define NEGS 0.01f

__device__ __forceinline__ float lrelu(float v) { return v > 0.f ? v : NEGS * v; }

// Async copy of one 16B chunk global -> LDS (GLOBAL_LOAD_ASYNC_TO_LDS_B128, GV mode).
__device__ __forceinline__ void async_g2l_b128(uint32_t lds_off, uint64_t gaddr) {
    asm volatile("global_load_async_to_lds_b128 %0, %1, off"
                 :: "v"(lds_off), "v"(gaddr) : "memory");
}
__device__ __forceinline__ void wait_async0() {
    asm volatile("s_wait_asynccnt 0" ::: "memory");
}

// ---------------- pack kernels ----------------
__global__ __launch_bounds__(256) void pack_wl_kernel(const float* __restrict__ Wl,
                                                      _Float16* __restrict__ WlB) {
    int idx = blockIdx.x * 256 + threadIdx.x;       // D*64*128 = 262144
    int k = idx & 127;
    int row = idx >> 7;                              // d*64 + o
    WlB[idx] = (k < 124) ? (_Float16)Wl[(size_t)row * 124 + k] : (_Float16)0.f;
}

__global__ __launch_bounds__(256) void pack_cast_kernel(const float* __restrict__ src,
                                                        _Float16* __restrict__ dst, int n) {
    int i = blockIdx.x * 256 + threadIdx.x;
    if (i < n) dst[i] = (_Float16)src[i];
}

// ---------------- stage 1: FE1 + grouped conv + pool -> feat16[R][128] ----------------
__global__ __launch_bounds__(256) void feat_kernel(const float* __restrict__ x,
                                                   const float* __restrict__ W1,
                                                   const float* __restrict__ b1,
                                                   const float* __restrict__ W2,
                                                   const float* __restrict__ b2,
                                                   _Float16* __restrict__ feat16) {
    int gid = blockIdx.x * 256 + threadIdx.x;   // over B*D*S, s fastest
    int s = gid & (NS - 1);
    int d = (gid >> 9) & (ND - 1);
    int b = gid >> 14;
    float xv = x[((size_t)b * ND + d) * NS + s];

    float h[64];
#pragma unroll
    for (int t = 0; t < 64; ++t)
        h[t] = lrelu(xv * W1[d * 64 + t] + b1[d * 64 + t]);

    size_t row = (size_t)b * NS + s;
    _Float16* out = feat16 + row * 128;
#pragma unroll
    for (int j = 0; j < 4; ++j) {
        float w0 = W2[(d * 4 + j) * 3 + 0];
        float w1 = W2[(d * 4 + j) * 3 + 1];
        float w2 = W2[(d * 4 + j) * 3 + 2];
        float bb = b2[d * 4 + j];
        float y[64];
#pragma unroll
        for (int t = 0; t < 64; ++t) {
            float hm = (t > 0)  ? h[t - 1] : 0.f;
            float hp = (t < 63) ? h[t + 1] : 0.f;
            y[t] = lrelu(w0 * hm + w1 * h[t] + w2 * hp + bb);
        }
#pragma unroll
        for (int i = 0; i < 31; ++i) {
            float v = fmaxf(fmaxf(y[2 * i], y[2 * i + 1]), y[2 * i + 2]);
            out[j * 31 + i] = (_Float16)v;
        }
    }
    out[124] = (_Float16)0.f; out[125] = (_Float16)0.f;
    out[126] = (_Float16)0.f; out[127] = (_Float16)0.f;
}

// ---------------- stage 2: per-dim Linear(124->64) via WMMA -> e16[(b,s,d)][64] ----------------
__global__ __launch_bounds__(128) void egemm_kernel(const _Float16* __restrict__ feat16, // [R][128]
                                                    const _Float16* __restrict__ WlB,    // [D][64][128]
                                                    const float* __restrict__ bl,        // [D][64]
                                                    _Float16* __restrict__ e16) {        // [R*D][64]
    __shared__ __align__(16) _Float16 atile[16 * 128];   // 4 KB A tile, shared by 4 waves
    const int tid  = threadIdx.x;
    const int lane = tid & 31;
    const int wave = tid >> 5;                // 0..3 -> N tile
    const int row0 = blockIdx.x * 16;
    const int d    = blockIdx.y;
    const int ml   = lane & 15;
    const int hi   = lane >> 4;
    const int n    = wave * 16 + ml;          // output column for B/D

    // async-stage the A tile once per block: 256 chunks of 16B, 2 per thread
    {
        uint32_t ldsBase = (uint32_t)(uintptr_t)atile;
#pragma unroll
        for (int c = 0; c < 2; ++c) {
            int chunk = tid + c * 128;                 // 0..255 = row*16 + seg
            uint64_t g = (uint64_t)(uintptr_t)(feat16 + (size_t)(row0 + (chunk >> 4)) * 128)
                       + (uint64_t)(chunk & 15) * 16;
            async_g2l_b128(ldsBase + (uint32_t)chunk * 16, g);
        }
        wait_async0();
    }
    __syncthreads();

    float bias = bl[d * 64 + n];
    v8f c = {};
#pragma unroll
    for (int r = 0; r < 8; ++r) c[r] = bias;

    const _Float16* brow = WlB + ((size_t)d * 64 + n) * 128;
    __builtin_prefetch(brow, 0, 1);

#pragma unroll
    for (int kk = 0; kk < 128; kk += 32) {
        V16H a, bm;
        a.q[0]  = *(const uint4*)(atile + ml * 128 + kk + 16 * hi);
        a.q[1]  = *(const uint4*)(atile + ml * 128 + kk + 16 * hi + 8);
        bm.q[0] = *(const uint4*)(brow + kk + 16 * hi);
        bm.q[1] = *(const uint4*)(brow + kk + 16 * hi + 8);
        c = __builtin_amdgcn_wmma_f32_16x16x32_f16(false, a.v, false, bm.v,
                                                   (short)0, c, false, false);
    }
#pragma unroll
    for (int r = 0; r < 8; ++r) {
        int mm = r + 8 * hi;
        e16[((size_t)(row0 + mm) * ND + d) * 64 + n] = (_Float16)lrelu(c[r]);
    }
}

// ---------------- generic gated-attention logits: tanh(A Wv^T+bv)*sig(A Wu^T+bu) . Wa + ba ----------
__global__ __launch_bounds__(256) void gated_logits_kernel(const _Float16* __restrict__ A16, // [M][64]
                                                           const _Float16* __restrict__ WvB, // [128][64]
                                                           const _Float16* __restrict__ WuB, // [128][64]
                                                           const float* __restrict__ bv,
                                                           const float* __restrict__ bu,
                                                           const float* __restrict__ Wa,     // [128]
                                                           const float* __restrict__ ba,
                                                           float* __restrict__ logits) {     // [M]
    __shared__ __align__(16) _Float16 atile[16 * 64];    // 2 KB A tile, shared by 8 waves
    __shared__ float part[8][16];
    const int tid  = threadIdx.x;
    const int lane = tid & 31;
    const int wave = tid >> 5;                // 0..7 -> N tile of 128
    const int row0 = blockIdx.x * 16;
    const int ml   = lane & 15;
    const int hi   = lane >> 4;
    const int n    = wave * 16 + ml;

    // async-stage A tile once per block: 128 chunks of 16B, first 4 waves
    if (tid < 128) {
        uint64_t g = (uint64_t)(uintptr_t)(A16 + (size_t)(row0 + (tid >> 3)) * 64)
                   + (uint64_t)(tid & 7) * 16;
        async_g2l_b128((uint32_t)(uintptr_t)atile + (uint32_t)tid * 16, g);
        wait_async0();
    }
    __syncthreads();

    v8f cv = {}, cu = {};
    const _Float16* vrow = WvB + (size_t)n * 64;
    const _Float16* urow = WuB + (size_t)n * 64;

#pragma unroll
    for (int kk = 0; kk < 64; kk += 32) {
        V16H a, bv16, bu16;
        a.q[0]    = *(const uint4*)(atile + ml * 64 + kk + 16 * hi);
        a.q[1]    = *(const uint4*)(atile + ml * 64 + kk + 16 * hi + 8);
        bv16.q[0] = *(const uint4*)(vrow + kk + 16 * hi);
        bv16.q[1] = *(const uint4*)(vrow + kk + 16 * hi + 8);
        bu16.q[0] = *(const uint4*)(urow + kk + 16 * hi);
        bu16.q[1] = *(const uint4*)(urow + kk + 16 * hi + 8);
        cv = __builtin_amdgcn_wmma_f32_16x16x32_f16(false, a.v, false, bv16.v,
                                                    (short)0, cv, false, false);
        cu = __builtin_amdgcn_wmma_f32_16x16x32_f16(false, a.v, false, bu16.v,
                                                    (short)0, cu, false, false);
    }

    float bvn = bv[n], bun = bu[n], wan = Wa[n];
    float tr[8];
#pragma unroll
    for (int r = 0; r < 8; ++r) {
        float av = tanhf(cv[r] + bvn);
        float au = 1.f / (1.f + __expf(-(cu[r] + bun)));
        tr[r] = av * au * wan;
    }
    // reduce over the 16 lanes sharing this m-group (xor masks < 16 stay inside group)
#pragma unroll
    for (int mask = 1; mask < 16; mask <<= 1)
#pragma unroll
        for (int r = 0; r < 8; ++r)
            tr[r] += __shfl_xor(tr[r], mask, 32);

    if (ml == 0) {
#pragma unroll
        for (int r = 0; r < 8; ++r) part[wave][r + 8 * hi] = tr[r];
    }
    __syncthreads();
    if (tid < 16) {
        float s = ba[0];
#pragma unroll
        for (int w = 0; w < 8; ++w) s += part[w][tid];
        logits[row0 + tid] = s;
    }
}

// ---------------- stage 4: softmax over D + feats = mean_d(e * a1) ----------------
__global__ __launch_bounds__(64) void attn1_kernel(const float* __restrict__ logits3,
                                                   const _Float16* __restrict__ e16,
                                                   float* __restrict__ a1,
                                                   float* __restrict__ feats32,
                                                   _Float16* __restrict__ feats16) {
    __shared__ float a1s[ND];
    int bs  = blockIdx.x;
    int tid = threadIdx.x;
    if (tid < 32) {   // wave 0: softmax over 32 dims
        float l = logits3[(size_t)bs * ND + tid];
        float mx = l;
#pragma unroll
        for (int m = 1; m < 32; m <<= 1) mx = fmaxf(mx, __shfl_xor(mx, m, 32));
        float e = __expf(l - mx);
        float sum = e;
#pragma unroll
        for (int m = 1; m < 32; m <<= 1) sum += __shfl_xor(sum, m, 32);
        float a = e / sum;
        a1[(size_t)bs * ND + tid] = a;
        a1s[tid] = a;
    }
    __syncthreads();
    int o = tid;   // 64 outputs
    float acc = 0.f;
#pragma unroll 4
    for (int dd = 0; dd < ND; ++dd)
        acc += (float)e16[((size_t)bs * ND + dd) * 64 + o] * a1s[dd];
    acc *= (1.f / (float)ND);
    feats32[(size_t)bs * 64 + o] = acc;
    feats16[(size_t)bs * 64 + o] = (_Float16)acc;
}

// ---------------- stage 6: softmax over S, pooled, LN head -> out1 ----------------
__global__ __launch_bounds__(256) void attn2_head_kernel(const float* __restrict__ logits2,
                                                         const float* __restrict__ feats32,
                                                         const float* __restrict__ Wm1,
                                                         const float* __restrict__ bm1,
                                                         const float* __restrict__ ln_g,
                                                         const float* __restrict__ ln_b,
                                                         const float* __restrict__ Wm2,
                                                         const float* __restrict__ bm2,
                                                         float* __restrict__ a2,
                                                         float* __restrict__ out1buf,
                                                         float* __restrict__ d_out) {
    __shared__ float red[256];
    __shared__ float a2s[NS];
    __shared__ float pooledp[4][64];
    __shared__ float pooled[64];
    int b = blockIdx.x, tid = threadIdx.x;
    const float* lg = logits2 + (size_t)b * NS;
    float v0 = lg[tid], v1 = lg[tid + 256];

    red[tid] = fmaxf(v0, v1);
    __syncthreads();
    for (int st = 128; st > 0; st >>= 1) { if (tid < st) red[tid] = fmaxf(red[tid], red[tid + st]); __syncthreads(); }
    float mx = red[0];
    __syncthreads();
    float e0 = __expf(v0 - mx), e1 = __expf(v1 - mx);
    red[tid] = e0 + e1;
    __syncthreads();
    for (int st = 128; st > 0; st >>= 1) { if (tid < st) red[tid] += red[tid + st]; __syncthreads(); }
    float inv = 1.f / red[0];
    a2s[tid] = e0 * inv;  a2s[tid + 256] = e1 * inv;
    a2[(size_t)b * NS + tid] = e0 * inv;
    a2[(size_t)b * NS + tid + 256] = e1 * inv;
    __syncthreads();

    int o = tid & 63, ch = tid >> 6;
    float acc = 0.f;
    for (int s = ch * 128; s < ch * 128 + 128; ++s)
        acc += feats32[((size_t)b * NS + s) * 64 + o] * a2s[s];
    pooledp[ch][o] = acc;
    __syncthreads();
    if (tid < 64)
        pooled[tid] = (pooledp[0][tid] + pooledp[1][tid] + pooledp[2][tid] + pooledp[3][tid]) * (1.f / (float)NS);
    __syncthreads();

    if (tid < 32) {
        float m = bm1[tid];
#pragma unroll 8
        for (int oo = 0; oo < 64; ++oo) m += pooled[oo] * Wm1[tid * 64 + oo];
        float mu = m;
#pragma unroll
        for (int msk = 1; msk < 32; msk <<= 1) mu += __shfl_xor(mu, msk, 32);
        mu *= (1.f / 32.f);
        float dv = m - mu;
        float var = dv * dv;
#pragma unroll
        for (int msk = 1; msk < 32; msk <<= 1) var += __shfl_xor(var, msk, 32);
        var *= (1.f / 32.f);
        float nrm = dv * rsqrtf(var + 1e-5f) * ln_g[tid] + ln_b[tid];
        float sc = lrelu(nrm) * Wm2[tid];
#pragma unroll
        for (int msk = 1; msk < 32; msk <<= 1) sc += __shfl_xor(sc, msk, 32);
        if (tid == 0) {
            float o1 = sc + bm2[0];
            o1 = o1 > 0.f ? o1 : 0.f;
            out1buf[b] = o1;
            d_out[b]   = o1;   // out1
        }
    }
}

// ---------------- stage 7: out2 / out3 expansion ----------------
__global__ __launch_bounds__(256) void out_kernel(const float* __restrict__ out1buf,
                                                  const float* __restrict__ a2,
                                                  const float* __restrict__ a1,
                                                  float* __restrict__ d_out) {
    int bs = blockIdx.x * 256 + threadIdx.x;    // 16384
    int b  = bs >> 9;
    float o1 = out1buf[b];
    float o2 = o1 * a2[bs];
    d_out[NB + bs] = o2;                        // out2 after 32 out1 floats
    float* o3 = d_out + NB + NR + (size_t)bs * ND;
#pragma unroll 4
    for (int dd = 0; dd < ND; ++dd) o3[dd] = o2 * a1[(size_t)bs * ND + dd];
}

// ---------------- launch ----------------
extern "C" void kernel_launch(void* const* d_in, const int* in_sizes, int n_in,
                              void* d_out, int out_size, void* d_ws, size_t ws_size,
                              hipStream_t stream) {
    const float* x   = (const float*)d_in[0];
    const float* W1  = (const float*)d_in[1];
    const float* b1  = (const float*)d_in[2];
    const float* W2  = (const float*)d_in[3];
    const float* b2  = (const float*)d_in[4];
    const float* Wl  = (const float*)d_in[5];
    const float* bl  = (const float*)d_in[6];
    const float* Wv  = (const float*)d_in[7];
    const float* bv  = (const float*)d_in[8];
    const float* Wu  = (const float*)d_in[9];
    const float* bu  = (const float*)d_in[10];
    const float* Wa  = (const float*)d_in[11];
    const float* ba  = (const float*)d_in[12];
    const float* Wv2 = (const float*)d_in[13];
    const float* bv2 = (const float*)d_in[14];
    const float* Wu2 = (const float*)d_in[15];
    const float* bu2 = (const float*)d_in[16];
    const float* Wa2 = (const float*)d_in[17];
    const float* ba2 = (const float*)d_in[18];
    const float* Wm1 = (const float*)d_in[19];
    const float* bm1 = (const float*)d_in[20];
    const float* lng = (const float*)d_in[21];
    const float* lnb = (const float*)d_in[22];
    const float* Wm2 = (const float*)d_in[23];
    const float* bm2 = (const float*)d_in[24];
    float* out = (float*)d_out;

    // carve workspace
    char* ws = (char*)d_ws;
    size_t off = 0;
    auto take = [&](size_t bytes) { void* p = ws + off; off = (off + bytes + 255) & ~(size_t)255; return p; };
    _Float16* feat16  = (_Float16*)take((size_t)NR * 128 * 2);       // 4 MB
    _Float16* WlB     = (_Float16*)take((size_t)ND * 64 * 128 * 2);  // 512 KB
    _Float16* WvB     = (_Float16*)take(128 * 64 * 2);
    _Float16* WuB     = (_Float16*)take(128 * 64 * 2);
    _Float16* Wv2B    = (_Float16*)take(128 * 64 * 2);
    _Float16* Wu2B    = (_Float16*)take(128 * 64 * 2);
    _Float16* e16     = (_Float16*)take((size_t)NM3 * 64 * 2);       // 64 MB
    float*    logits3 = (float*)take((size_t)NM3 * 4);               // 2 MB
    float*    a1buf   = (float*)take((size_t)NM3 * 4);               // 2 MB
    float*    feats32 = (float*)take((size_t)NR * 64 * 4);           // 4 MB
    _Float16* feats16 = (_Float16*)take((size_t)NR * 64 * 2);        // 2 MB
    float*    logits2 = (float*)take((size_t)NR * 4);
    float*    a2buf   = (float*)take((size_t)NR * 4);
    float*    out1buf = (float*)take(NB * 4);
    (void)ws_size; (void)in_sizes; (void)n_in; (void)out_size;

    // weight packing
    pack_wl_kernel<<<(ND * 64 * 128) / 256, 256, 0, stream>>>(Wl, WlB);
    pack_cast_kernel<<<(128 * 64 + 255) / 256, 256, 0, stream>>>(Wv,  WvB,  128 * 64);
    pack_cast_kernel<<<(128 * 64 + 255) / 256, 256, 0, stream>>>(Wu,  WuB,  128 * 64);
    pack_cast_kernel<<<(128 * 64 + 255) / 256, 256, 0, stream>>>(Wv2, Wv2B, 128 * 64);
    pack_cast_kernel<<<(128 * 64 + 255) / 256, 256, 0, stream>>>(Wu2, Wu2B, 128 * 64);

    // stage 1: features
    feat_kernel<<<(NB * ND * NS) / 256, 256, 0, stream>>>(x, W1, b1, W2, b2, feat16);

    // stage 2: per-dim Linear via WMMA
    egemm_kernel<<<dim3(NR / 16, ND), 128, 0, stream>>>(feat16, WlB, bl, e16);

    // stage 3: gated attention over dims -> logits3
    gated_logits_kernel<<<NM3 / 16, 256, 0, stream>>>(e16, WvB, WuB, bv, bu, Wa, ba, logits3);

    // stage 4: softmax over D + feats
    attn1_kernel<<<NR, 64, 0, stream>>>(logits3, e16, a1buf, feats32, feats16);

    // stage 5: gated attention over sub-windows -> logits2
    gated_logits_kernel<<<NR / 16, 256, 0, stream>>>(feats16, Wv2B, Wu2B, bv2, bu2, Wa2, ba2, logits2);

    // stage 6: softmax over S, pooled, head -> out1
    attn2_head_kernel<<<NB, 256, 0, stream>>>(logits2, feats32, Wm1, bm1, lng, lnb, Wm2, bm2,
                                              a2buf, out1buf, out);

    // stage 7: out2 / out3
    out_kernel<<<NR / 256, 256, 0, stream>>>(out1buf, a2buf, a1buf, out);
}
